// GCN_13159779795712
// MI455X (gfx1250) — matrix operand order repaired and checked by reading, not compiled, not run on hardware
//
#include <hip/hip_runtime.h>

#define NN   50000
#define EE   1600000
#define DIN  128
#define HH   64
#define CC   32
#define NIDX 10000
#define EPSV 1e-5f

typedef __attribute__((ext_vector_type(16))) _Float16 v16h;
typedef __attribute__((ext_vector_type(8)))  float    v8f;

union H2U { unsigned u; _Float16 h[2]; };

__device__ __forceinline__ v8f wmma_f16(v16h a, v16h b, v8f c) {
  // (neg_a, A, neg_b, B, c_mod, C, reuse_a, reuse_b)
  return __builtin_amdgcn_wmma_f32_16x16x32_f16(false, a, false, b, (short)0, c,
                                                false, false);
}

// ---------------------------------------------------------------------------
// GEMM1: out[N,64] = features[N,128] @ W1[128,64] + b1      (WMMA f16, acc f32)
// 128 threads = 4 waves; each wave computes one 16x64 tile.
// ---------------------------------------------------------------------------
__global__ void __launch_bounds__(128)
k_gemm1(const float* __restrict__ F, const float* __restrict__ W1,
        const float* __restrict__ b1, float* __restrict__ out)
{
  __shared__ __align__(16) _Float16 wt[HH * DIN];   // W1^T, [n][k], f16 (16 KB)
  int tid = threadIdx.x;
  for (int i = tid; i < HH * DIN; i += 128) {       // coalesced W1 read
    int n = i & 63, k = i >> 6;
    wt[n * DIN + k] = (_Float16)W1[i];
  }
  __syncthreads();

  int lane = tid & 31, wv = tid >> 5;
  int tile = blockIdx.x * 4 + wv;                   // 16-row output tile
  if (tile >= NN / 16) return;                      // wave-uniform guard

  int nloc = lane & 15;
  int hA   = (lane < 16) ? 0 : 8;                   // A K-half select / C M-half
  int hB   = (lane < 16) ? 0 : 16;                  // B K-half select
  const float* fr = F + (size_t)(tile * 16 + nloc) * DIN;   // A row per lane

  v8f acc[4] = {};
  for (int k0 = 0; k0 < DIN; k0 += 32) {
    v16h a;
#pragma unroll
    for (int v = 0; v < 8; ++v) {
      int kl = k0 + 2 * hA + 2 * v;
      a[2 * v]     = (_Float16)fr[kl];
      a[2 * v + 1] = (_Float16)fr[kl + 1];
    }
#pragma unroll
    for (int t = 0; t < 4; ++t) {
      const unsigned* wp =
          (const unsigned*)(wt + (nloc + t * 16) * DIN + k0 + hB);
      v16h b;
#pragma unroll
      for (int v = 0; v < 8; ++v) {
        H2U cv; cv.u = wp[v];
        b[2 * v] = cv.h[0]; b[2 * v + 1] = cv.h[1];
      }
      acc[t] = wmma_f16(a, b, acc[t]);
    }
  }

  int mbase = tile * 16 + hA;
#pragma unroll
  for (int t = 0; t < 4; ++t) {
    float bv = b1[nloc + t * 16];
#pragma unroll
    for (int r = 0; r < 8; ++r)
      out[(size_t)(mbase + r) * HH + t * 16 + nloc] = acc[t][r] + bv;
  }
}

// ---------------------------------------------------------------------------
// SPMM init: h[n][k] = c[k]  (folds the +c bias) and zero BN stats slot.
// ---------------------------------------------------------------------------
__global__ void __launch_bounds__(256)
k_spmm_init(float* __restrict__ Hbuf, const float* __restrict__ c,
            float* __restrict__ stats)
{
  size_t gid = (size_t)blockIdx.x * 256 + threadIdx.x;
  if (gid < 128) stats[gid] = 0.0f;
  size_t total  = (size_t)NN * HH;
  size_t stride = (size_t)gridDim.x * 256;
  for (size_t i = gid; i < total; i += stride) Hbuf[i] = c[i & 63];
}

// ---------------------------------------------------------------------------
// SPMM: h[row] += val * x[col]   — 4 threads per edge, 16 features each.
// x and h (12.8 MB each) stay resident in the 192 MB L2; atomics are
// non-returning global_atomic_add_f32 that resolve in L2.
// ---------------------------------------------------------------------------
__global__ void __launch_bounds__(256)
k_spmm(const int* __restrict__ ai, const float* __restrict__ av,
       const float* __restrict__ X, float* __restrict__ Hbuf)
{
  int gid = blockIdx.x * 256 + threadIdx.x;
  if (gid >= EE * 4) return;
  int e  = gid >> 2;
  int c4 = (gid & 3) * 16;
  __builtin_prefetch(ai + 2 * e + 1024, 0, 0);      // stream edge list ahead
  int   r   = ai[2 * e];
  int   cix = ai[2 * e + 1];
  float v   = av[e];
  const float4* xp = (const float4*)(X + (size_t)cix * HH + c4);
  float*        hp = Hbuf + (size_t)r * HH + c4;
#pragma unroll
  for (int q = 0; q < 4; ++q) {
    float4 x4 = xp[q];
    atomicAdd(hp + 4 * q + 0, v * x4.x);
    atomicAdd(hp + 4 * q + 1, v * x4.y);
    atomicAdd(hp + 4 * q + 2, v * x4.z);
    atomicAdd(hp + 4 * q + 3, v * x4.w);
  }
}

// ---------------------------------------------------------------------------
// Column statistics: stats[k] = sum, stats[64+k] = sumsq over N rows.
// ---------------------------------------------------------------------------
__global__ void __launch_bounds__(256)
k_colstats(const float* __restrict__ Hbuf, float* __restrict__ stats)
{
  __shared__ float ls[256], ls2[256];
  int tid = threadIdx.x;
  int k   = tid & 63;
  int sub = tid >> 6;                 // 0..3 row sub-lanes per block
  float s = 0.f, s2 = 0.f;
  for (int r = blockIdx.x * 4 + sub; r < NN; r += gridDim.x * 4) {
    float x = Hbuf[(size_t)r * HH + k];
    s += x; s2 += x * x;
  }
  ls[tid] = s; ls2[tid] = s2;
  __syncthreads();
  if (tid < 64) {
    s  = ls[tid]  + ls[tid + 64]  + ls[tid + 128]  + ls[tid + 192];
    s2 = ls2[tid] + ls2[tid + 64] + ls2[tid + 128] + ls2[tid + 192];
    atomicAdd(&stats[k], s);
    atomicAdd(&stats[64 + k], s2);
  }
}

// ---------------------------------------------------------------------------
// Finalize BN: precompute scale = gamma*rsqrt(var+eps), shift = beta - mean*scale
// so BN+ReLU fuses into the next GEMM's A-operand load.
// ---------------------------------------------------------------------------
__global__ void __launch_bounds__(64)
k_finalize(const float* __restrict__ stats, const float* __restrict__ gamma,
           const float* __restrict__ beta, float* __restrict__ sc,
           float* __restrict__ sh)
{
  int k = threadIdx.x;
  float inv  = 1.0f / (float)NN;
  float mean = stats[k] * inv;
  float var  = stats[64 + k] * inv - mean * mean;
  float rs   = rsqrtf(var + EPSV);
  float s    = rs * gamma[k];
  sc[k] = s;
  sh[k] = beta[k] - mean * s;
}

// ---------------------------------------------------------------------------
// GEMM2: out[N,64] = relu(BN(h)) @ W2[64,64]   (BN+ReLU fused into A load)
// ---------------------------------------------------------------------------
__global__ void __launch_bounds__(128)
k_gemm2(const float* __restrict__ Hin, const float* __restrict__ W2,
        const float* __restrict__ sc, const float* __restrict__ sh,
        float* __restrict__ out)
{
  __shared__ __align__(16) _Float16 wt[HH * HH];    // W2^T f16 (8 KB)
  __shared__ float ssc[HH], ssh[HH];
  int tid = threadIdx.x;
  for (int i = tid; i < HH * HH; i += 128) {
    int n = i & 63, k = i >> 6;
    wt[n * HH + k] = (_Float16)W2[i];
  }
  if (tid < 64) { ssc[tid] = sc[tid]; ssh[tid] = sh[tid]; }
  __syncthreads();

  int lane = tid & 31, wv = tid >> 5;
  int tile = blockIdx.x * 4 + wv;
  if (tile >= NN / 16) return;

  int nloc = lane & 15;
  int hA   = (lane < 16) ? 0 : 8;
  int hB   = (lane < 16) ? 0 : 16;
  const float* fr = Hin + (size_t)(tile * 16 + nloc) * HH;

  v8f acc[4] = {};
  for (int k0 = 0; k0 < HH; k0 += 32) {
    v16h a;
#pragma unroll
    for (int v = 0; v < 8; ++v) {
      int kl = k0 + 2 * hA + 2 * v;
      a[2 * v]     = (_Float16)fmaxf(fr[kl] * ssc[kl] + ssh[kl], 0.0f);
      a[2 * v + 1] = (_Float16)fmaxf(fr[kl + 1] * ssc[kl + 1] + ssh[kl + 1], 0.0f);
    }
#pragma unroll
    for (int t = 0; t < 4; ++t) {
      const unsigned* wp =
          (const unsigned*)(wt + (nloc + t * 16) * HH + k0 + hB);
      v16h b;
#pragma unroll
      for (int v = 0; v < 8; ++v) {
        H2U cv; cv.u = wp[v];
        b[2 * v] = cv.h[0]; b[2 * v + 1] = cv.h[1];
      }
      acc[t] = wmma_f16(a, b, acc[t]);
    }
  }

  int mbase = tile * 16 + hA;
#pragma unroll
  for (int t = 0; t < 4; ++t)
#pragma unroll
    for (int r = 0; r < 8; ++r)
      out[(size_t)(mbase + r) * HH + t * 16 + nloc] = acc[t][r];
}

// ---------------------------------------------------------------------------
// Output GEMM with gather: out[i,:] = relu(BN2(h[idx[i]])) @ Wout + bout
// 10000 rows = 625 exact 16-row tiles; two 16-wide N-tiles (C=32).
// ---------------------------------------------------------------------------
__global__ void __launch_bounds__(128)
k_gemm_out(const float* __restrict__ Hin, const int* __restrict__ idx,
           const float* __restrict__ Wout, const float* __restrict__ bout,
           const float* __restrict__ sc, const float* __restrict__ sh,
           float* __restrict__ out)
{
  __shared__ __align__(16) _Float16 wt[CC * HH];    // Wout^T f16 (4 KB)
  __shared__ float ssc[HH], ssh[HH];
  int tid = threadIdx.x;
  for (int i = tid; i < CC * HH; i += 128) {
    int n = i & 31, k = i >> 5;
    wt[n * HH + k] = (_Float16)Wout[i];
  }
  if (tid < 64) { ssc[tid] = sc[tid]; ssh[tid] = sh[tid]; }
  __syncthreads();

  int lane = tid & 31, wv = tid >> 5;
  int tile = blockIdx.x * 4 + wv;
  if (tile >= NIDX / 16) return;

  int nloc = lane & 15;
  int hA   = (lane < 16) ? 0 : 8;
  int hB   = (lane < 16) ? 0 : 16;
  int row  = idx[tile * 16 + nloc];
  const float* fr = Hin + (size_t)row * HH;

  v8f acc[2] = {};
  for (int k0 = 0; k0 < HH; k0 += 32) {
    v16h a;
#pragma unroll
    for (int v = 0; v < 8; ++v) {
      int kl = k0 + 2 * hA + 2 * v;
      a[2 * v]     = (_Float16)fmaxf(fr[kl] * ssc[kl] + ssh[kl], 0.0f);
      a[2 * v + 1] = (_Float16)fmaxf(fr[kl + 1] * ssc[kl + 1] + ssh[kl + 1], 0.0f);
    }
#pragma unroll
    for (int t = 0; t < 2; ++t) {
      const unsigned* wp =
          (const unsigned*)(wt + (nloc + t * 16) * HH + k0 + hB);
      v16h b;
#pragma unroll
      for (int v = 0; v < 8; ++v) {
        H2U cv; cv.u = wp[v];
        b[2 * v] = cv.h[0]; b[2 * v + 1] = cv.h[1];
      }
      acc[t] = wmma_f16(a, b, acc[t]);
    }
  }

  int mbase = tile * 16 + hA;
#pragma unroll
  for (int t = 0; t < 2; ++t) {
    float bv = bout[nloc + t * 16];
#pragma unroll
    for (int r = 0; r < 8; ++r)
      out[(size_t)(mbase + r) * CC + t * 16 + nloc] = acc[t][r] + bv;
  }
}

// ---------------------------------------------------------------------------
extern "C" void kernel_launch(void* const* d_in, const int* in_sizes, int n_in,
                              void* d_out, int out_size, void* d_ws, size_t ws_size,
                              hipStream_t stream)
{
  const float* features = (const float*)d_in[0];
  const int*   adj_i    = (const int*)  d_in[1];
  const float* adj_v    = (const float*)d_in[2];
  const int*   idx      = (const int*)  d_in[3];
  const float* W1   = (const float*)d_in[4];
  const float* b1   = (const float*)d_in[5];
  const float* c1   = (const float*)d_in[6];
  const float* g1   = (const float*)d_in[7];
  const float* be1  = (const float*)d_in[8];
  const float* W2   = (const float*)d_in[9];
  const float* c2   = (const float*)d_in[10];
  const float* g2   = (const float*)d_in[11];
  const float* be2  = (const float*)d_in[12];
  const float* Wo   = (const float*)d_in[13];
  const float* bo   = (const float*)d_in[14];

  float* bufA  = (float*)d_ws;                       // x after dense GEMM
  float* bufB  = bufA + (size_t)NN * HH;             // h after SPMM
  float* stats = bufB + (size_t)NN * HH;             // 128 floats (sum, sumsq)
  float* sc1 = stats + 128; float* sh1 = sc1 + 64;
  float* sc2 = sh1 + 64;    float* sh2 = sc2 + 64;

  const int gemm_blocks = (NN / 16 + 3) / 4;         // 782
  const int spmm_blocks = (EE * 4) / 256;            // 25000
  const int out_blocks  = (NIDX / 16 + 3) / 4;       // 157

  // Layer 1
  k_gemm1<<<gemm_blocks, 128, 0, stream>>>(features, W1, b1, bufA);
  k_spmm_init<<<2048, 256, 0, stream>>>(bufB, c1, stats);
  k_spmm<<<spmm_blocks, 256, 0, stream>>>(adj_i, adj_v, bufA, bufB);
  k_colstats<<<512, 256, 0, stream>>>(bufB, stats);
  k_finalize<<<1, 64, 0, stream>>>(stats, g1, be1, sc1, sh1);

  // Layer 2
  k_gemm2<<<gemm_blocks, 128, 0, stream>>>(bufB, W2, sc1, sh1, bufA);
  k_spmm_init<<<2048, 256, 0, stream>>>(bufB, c2, stats);
  k_spmm<<<spmm_blocks, 256, 0, stream>>>(adj_i, adj_v, bufA, bufB);
  k_colstats<<<512, 256, 0, stream>>>(bufB, stats);
  k_finalize<<<1, 64, 0, stream>>>(stats, g2, be2, sc2, sh2);

  // Output projection + gather
  k_gemm_out<<<out_blocks, 128, 0, stream>>>(bufB, idx, Wo, bo, sc2, sh2,
                                             (float*)d_out);
}
